// SeqGraphRepNetwork_26070451486836
// MI455X (gfx1250) — matrix operand order repaired and checked by reading, not compiled, not run on hardware
//
#include <hip/hip_runtime.h>

typedef _Float16 half_t;
typedef __attribute__((ext_vector_type(16))) _Float16 v16h;
typedef __attribute__((ext_vector_type(8)))  float    v8f;

#define D128   128
#define LMAX   160
#define NHEAD  4
#define HDIM   32

// ---------------------------------------------------------------- helpers

union V16H { unsigned u[8]; _Float16 h[16]; v16h v; };

__device__ __forceinline__ v8f zero8() {
    v8f z = {0.f,0.f,0.f,0.f,0.f,0.f,0.f,0.f};
    return z;
}

__device__ __forceinline__ v8f wmma32(v16h a, v16h b, v8f c) {
    // D(f32 16x16) = A(f16 16x32) * B(f16 32x16) + C
    return __builtin_amdgcn_wmma_f32_16x16x32_f16(false, a, false, b,
                                                  (short)0, c, false, false);
}

// CDNA5 async direct-to-LDS copy (GLOBAL_LOAD_ASYNC_TO_LDS_B128, ASYNCcnt).
// 16 bytes per lane; LDS dest VGPR holds the LDS byte address (low 32 bits of
// the generic pointer), 64-bit global address in a VGPR pair, GV mode.
__device__ __forceinline__ void async_copy_b128(half_t* lds_dst, const half_t* gsrc) {
    unsigned ldsa = (unsigned)(size_t)(void*)lds_dst;
    unsigned long long ga = (unsigned long long)(size_t)gsrc;
    asm volatile("global_load_async_to_lds_b128 %0, %1, off"
                 :: "v"(ldsa), "v"(ga)
                 : "memory");
}
__device__ __forceinline__ void wait_async0() {
    asm volatile("s_wait_asynccnt 0x0" ::: "memory");
}

// A-matrix 16x32 f16 loader from a row-major source (row stride `ld` halves).
// Contiguous K order (matches B loaders): lane<16 row M=lane holds K 0..15; lane>=16 holds K 16..31.
__device__ __forceinline__ v16h load_a(const half_t* base, int ld) {
    int lane = threadIdx.x & 31;
    int m = lane & 15, hi = lane >> 4;
    const half_t* row = base + m * ld;
    V16H r;
#pragma unroll
    for (int v = 0; v < 8; ++v) {
        int k0 = hi * 16 + 2 * v;
        r.u[v] = *(const unsigned*)(row + k0);
    }
    return r.v;
}

// B-matrix 32x16 f16 loader where B[k][n] = src[n*ld + k] (i.e. X @ W^T with W
// row-major, or Q @ K^T with K rows). Contiguous K order: lane = column n, lanes<16 hold
// K=0..15 (2 per VGPR), lanes>=16 hold K=16..31.
__device__ __forceinline__ v16h load_b_rows(const half_t* base, int ld) {
    int lane = threadIdx.x & 31;
    int n = lane & 15, hi = lane >> 4;
    const half_t* row = base + n * ld;
    V16H r;
#pragma unroll
    for (int v = 0; v < 8; ++v) {
        int k0 = hi * 16 + 2 * v;
        r.u[v] = *(const unsigned*)(row + k0);
    }
    return r.v;
}

// B-matrix 32x16 f16 loader where B[k][n] = src[k*ld + n] (true column read,
// used for P @ V where V rows are k-positions).
__device__ __forceinline__ v16h load_b_cols(const half_t* base, int ld) {
    int lane = threadIdx.x & 31;
    int n = lane & 15, hi = lane >> 4;
    V16H r;
#pragma unroll
    for (int v = 0; v < 8; ++v) {
        int k = hi * 16 + 2 * v;
        r.h[2 * v]     = base[(k) * ld + n];
        r.h[2 * v + 1] = base[(k + 1) * ld + n];
    }
    return r.v;
}

__device__ __forceinline__ float redsum32(float v) {
#pragma unroll
    for (int m = 16; m >= 1; m >>= 1) v += __shfl_xor(v, m, 32);
    return v;
}
__device__ __forceinline__ float redsum16(float v) {
#pragma unroll
    for (int m = 8; m >= 1; m >>= 1) v += __shfl_xor(v, m, 32);
    return v;
}
__device__ __forceinline__ float redmax16(float v) {
#pragma unroll
    for (int m = 8; m >= 1; m >>= 1) v = fmaxf(v, __shfl_xor(v, m, 32));
    return v;
}

// ---------------------------------------------------------------- kernel 1: prep
// w_src = att_W^T @ a_src ; w_dst = att_W^T @ a_dst ; starts = cumsum(lengths)
__global__ void k_prep(const float* __restrict__ attW,
                       const float* __restrict__ a_src,
                       const float* __restrict__ a_dst,
                       const int*  __restrict__ lengths,
                       float* __restrict__ w_src, float* __restrict__ w_dst,
                       int* __restrict__ starts, int B) {
    int t = threadIdx.x;
    if (t < D128) {
        float s = 0.f, d = 0.f;
        for (int k = 0; k < D128; ++k) {
            float w = attW[k * D128 + t];
            s += w * a_src[k];
            d += w * a_dst[k];
        }
        w_src[t] = s;
        w_dst[t] = d;
    }
    if (t == 0) {
        int acc = 0;
        for (int b = 0; b < B; ++b) { starts[b] = acc; acc += lengths[b]; }
        starts[B] = acc;
    }
}

// ---------------------------------------------------------------- kernel 2: GCN
// One block per session; one wave per node. Each node has <=2 incoming msgs:
//   forward  (pred -> n): logit = dot(x[n] + edge_emb, w_src), payload x[pred]
//   backward (succ -> n): logit = dot(x[n], w_dst),            payload x[succ]
__global__ __launch_bounds__(256) void k_gcn(
        const float* __restrict__ POI, const float* __restrict__ dde,
        const int* __restrict__ sess_idx, const int* __restrict__ edge_dist,
        const int* __restrict__ lengths, const int* __restrict__ starts,
        const float* __restrict__ w_src, const float* __restrict__ w_dst,
        half_t* __restrict__ Hh) {
    int b = blockIdx.x;
    int L = lengths[b], s0 = starts[b];
    int eb = s0 - b;  // first edge index of this session
    half_t* Hs = Hh + (size_t)b * LMAX * D128;
    for (int i = threadIdx.x; i < LMAX * D128 / 2; i += 256)
        ((unsigned*)Hs)[i] = 0u;
    __syncthreads();

    int wave = threadIdx.x >> 5, lane = threadIdx.x & 31;
    float ws4[4], wd4[4];
#pragma unroll
    for (int j = 0; j < 4; ++j) {
        ws4[j] = w_src[lane * 4 + j];
        wd4[j] = w_dst[lane * 4 + j];
    }
    for (int p = wave; p < L; p += 8) {
        int n = s0 + p;
        const float* xn = POI + (size_t)sess_idx[n] * D128;
        float x4[4];
#pragma unroll
        for (int j = 0; j < 4; ++j) x4[j] = xn[lane * 4 + j];
        bool hasF = (p > 0), hasB = (p < L - 1);
        float lf = -3.0e38f, lb = -3.0e38f;
        if (hasB) {
            float t = 0.f;
#pragma unroll
            for (int j = 0; j < 4; ++j) t += x4[j] * wd4[j];
            lb = redsum32(t);
        }
        if (hasF) {
            const float* el = dde + (size_t)edge_dist[eb + p - 1] * D128;
            float t = 0.f;
#pragma unroll
            for (int j = 0; j < 4; ++j) t += (x4[j] + el[lane * 4 + j]) * ws4[j];
            lf = redsum32(t);
        }
        float m  = fmaxf(lf, lb);
        float ef = hasF ? __expf(lf - m) : 0.f;
        float eg = hasB ? __expf(lb - m) : 0.f;
        float inv = 1.f / (ef + eg + 1e-16f);
        float af = ef * inv, ab = eg * inv;
        float h4[4] = {0.f, 0.f, 0.f, 0.f};
        if (hasF) {
            const float* xp = POI + (size_t)sess_idx[n - 1] * D128;
#pragma unroll
            for (int j = 0; j < 4; ++j) h4[j] += af * xp[lane * 4 + j];
        }
        if (hasB) {
            const float* xs = POI + (size_t)sess_idx[n + 1] * D128;
#pragma unroll
            for (int j = 0; j < 4; ++j) h4[j] += ab * xs[lane * 4 + j];
        }
#pragma unroll
        for (int j = 0; j < 4; ++j)
            Hs[p * D128 + lane * 4 + j] = (half_t)h4[j];
    }
}

// ---------------------------------------------------------------- kernel 3: K/V projection
// grid = (row_tiles/8, 2): y==0 -> k = H@Wk^T+bk, y==1 -> v = H@Wv^T+bv
__global__ __launch_bounds__(256) void k_kv(
        const half_t* __restrict__ Hh, const float* __restrict__ in_proj_w,
        const float* __restrict__ in_proj_b,
        half_t* __restrict__ kh, half_t* __restrict__ vh, int Mtiles) {
    __shared__ __align__(16) half_t wsm[D128 * D128];
    __shared__ float bs[D128];
    int which = blockIdx.y;
    const float* wsrc = in_proj_w + (size_t)(which + 1) * D128 * D128;
    const float* bsrc = in_proj_b + (which + 1) * D128;
    int tid = threadIdx.x;
    for (int i = tid; i < D128 * D128; i += 256) wsm[i] = (half_t)wsrc[i];
    if (tid < D128) bs[tid] = bsrc[tid];
    __syncthreads();

    int wave = tid >> 5, lane = tid & 31, m15 = lane & 15, hi = lane >> 4;
    int rt = blockIdx.x * 8 + wave;
    if (rt >= Mtiles) return;
    const half_t* abase = Hh + (size_t)rt * 16 * D128;
    half_t* dst = (which ? vh : kh) + (size_t)rt * 16 * D128;
#pragma unroll
    for (int n = 0; n < 8; ++n) {
        v8f acc = zero8();
#pragma unroll
        for (int kc = 0; kc < 4; ++kc) {
            v16h a  = load_a(abase + kc * 32, D128);
            v16h bm = load_b_rows(wsm + (n * 16) * D128 + kc * 32, D128);
            acc = wmma32(a, bm, acc);
        }
        float bias = bs[n * 16 + m15];
#pragma unroll
        for (int r = 0; r < 8; ++r)
            dst[(r + 8 * hi) * D128 + n * 16 + m15] = (half_t)(acc[r] + bias);
    }
}

// ---------------------------------------------------------------- kernel 4: fused attention
// Per session: LN1 -> q-proj -> masked 2-pass softmax attention -> out-proj
// + residual(Q) -> LN2 -> Oh (f16). K/V and weights resident in LDS.
// K/V staged with CDNA5 async direct-to-LDS copies (ASYNCcnt), overlapped with
// the f32->f16 weight conversion done through VGPRs.
__global__ __launch_bounds__(256) void k_attn(
        const half_t* __restrict__ Hh, const half_t* __restrict__ kh,
        const half_t* __restrict__ vh,
        const float* __restrict__ in_proj_w, const float* __restrict__ in_proj_b,
        const float* __restrict__ out_proj_w, const float* __restrict__ out_proj_b,
        const float* __restrict__ ln1g, const float* __restrict__ ln1b,
        const float* __restrict__ ln2g, const float* __restrict__ ln2b,
        const int* __restrict__ lengths, half_t* __restrict__ Oh) {
    __shared__ __align__(16) half_t kbuf[LMAX * D128];
    __shared__ __align__(16) half_t vbuf[(LMAX + 16) * D128];  // padded rows = 0
    __shared__ __align__(16) half_t wqs[D128 * D128];
    __shared__ __align__(16) half_t wos[D128 * D128];
    __shared__ __align__(16) half_t qln[8][16 * D128];   // LN1 tile (residual)
    __shared__ __align__(16) half_t qbuf[8][16 * D128];  // q tile, reused as ctx
    __shared__ __align__(16) half_t ptile[8][16 * 32];   // probs (A layout src)
    __shared__ float bqs[D128], bos[D128], g1[D128], bb1[D128], g2[D128], bb2[D128];

    int b = blockIdx.x, tid = threadIdx.x;
    int L = lengths[b];
    int wave = tid >> 5, lane = tid & 31, m15 = lane & 15, hi = lane >> 4;

    // ---- async stage K/V directly into LDS (no VGPR round trip)
    const half_t* kg = kh + (size_t)b * LMAX * D128;
    const half_t* vg = vh + (size_t)b * LMAX * D128;
#pragma unroll 2
    for (int i = tid; i < LMAX * D128 / 8; i += 256) {
        async_copy_b128(kbuf + i * 8, kg + i * 8);
        async_copy_b128(vbuf + i * 8, vg + i * 8);
    }
    // zero the 16 padded v rows (regular ds stores)
    for (int i = tid; i < 16 * D128 / 4; i += 256)
        ((uint2*)vbuf)[LMAX * 32 + i] = make_uint2(0u, 0u);
    // overlap: convert weights f32 -> f16 through VGPRs while asyncs fly
    for (int i = tid; i < D128 * D128; i += 256) {
        wqs[i] = (half_t)in_proj_w[i];  // Wq = rows 0..127
        wos[i] = (half_t)out_proj_w[i];
    }
    if (tid < D128) {
        bqs[tid] = in_proj_b[tid];  bos[tid] = out_proj_b[tid];
        g1[tid] = ln1g[tid];  bb1[tid] = ln1b[tid];
        g2[tid] = ln2g[tid];  bb2[tid] = ln2b[tid];
    }
    wait_async0();       // this wave's async-to-LDS copies complete
    __syncthreads();     // publish LDS across all waves

    const half_t* hg = Hh + (size_t)b * LMAX * D128;
    half_t* og = Oh + (size_t)b * LMAX * D128;
    int nk = (L + 15) >> 4;
    const float qscale = 0.17677669529663687f;  // 1/sqrt(32)

    for (int t = wave; t < LMAX / 16; t += 8) {
        int r0 = t * 16;
        if (r0 >= L) {  // fully padded tile: emit zeros
            for (int i = lane; i < 16 * D128 / 2; i += 32)
                ((unsigned*)(og + r0 * D128))[i] = 0u;
            continue;
        }
        // ---- LN1 of 16 H rows -> qln
        half_t* ql = qln[wave];
        for (int r = 0; r < 16; ++r) {
            const half_t* hr = hg + (r0 + r) * D128;
            float xv[4];
#pragma unroll
            for (int j = 0; j < 4; ++j) xv[j] = (float)hr[lane * 4 + j];
            float mu = redsum32(xv[0] + xv[1] + xv[2] + xv[3]) * (1.f / 128.f);
            float vs = 0.f;
#pragma unroll
            for (int j = 0; j < 4; ++j) { float d = xv[j] - mu; vs += d * d; }
            float rstd = rsqrtf(redsum32(vs) * (1.f / 128.f) + 1e-8f);
#pragma unroll
            for (int j = 0; j < 4; ++j) {
                int c = lane * 4 + j;
                ql[r * D128 + c] = (half_t)((xv[j] - mu) * rstd * g1[c] + bb1[c]);
            }
        }
        // ---- q = (LN1 @ Wq^T + bq) * qscale -> qbuf
        half_t* qb = qbuf[wave];
#pragma unroll
        for (int n = 0; n < 8; ++n) {
            v8f acc = zero8();
#pragma unroll
            for (int kc = 0; kc < 4; ++kc) {
                v16h a  = load_a(ql + kc * 32, D128);
                v16h bm = load_b_rows(wqs + (n * 16) * D128 + kc * 32, D128);
                acc = wmma32(a, bm, acc);
            }
            float bias = bqs[n * 16 + m15];
#pragma unroll
            for (int r = 0; r < 8; ++r)
                qb[(r + 8 * hi) * D128 + n * 16 + m15] =
                    (half_t)((acc[r] + bias) * qscale);
        }
        // ---- attention, head by head (2-pass masked softmax)
        half_t* pt = ptile[wave];
        for (int h = 0; h < NHEAD; ++h) {
            v16h aq = load_a(qb + h * HDIM, D128);
            float m8[8];
#pragma unroll
            for (int r = 0; r < 8; ++r) m8[r] = -3.0e38f;
            for (int kt = 0; kt < nk; ++kt) {
                v16h bk = load_b_rows(kbuf + (kt * 16) * D128 + h * HDIM, D128);
                v8f c = wmma32(aq, bk, zero8());
                bool valid = (kt * 16 + m15) < L;
#pragma unroll
                for (int r = 0; r < 8; ++r)
                    m8[r] = fmaxf(m8[r], valid ? c[r] : -3.0e38f);
            }
#pragma unroll
            for (int r = 0; r < 8; ++r) m8[r] = redmax16(m8[r]);

            float l8[8];
#pragma unroll
            for (int r = 0; r < 8; ++r) l8[r] = 0.f;
            v8f o0 = zero8(), o1 = zero8();
            for (int kt2 = 0; kt2 < nk; kt2 += 2) {
#pragma unroll
                for (int sub = 0; sub < 2; ++sub) {
                    int kt = kt2 + sub;
                    if (kt < nk) {
                        v16h bk = load_b_rows(kbuf + (kt * 16) * D128 + h * HDIM, D128);
                        v8f c = wmma32(aq, bk, zero8());
                        bool valid = (kt * 16 + m15) < L;
#pragma unroll
                        for (int r = 0; r < 8; ++r) {
                            float ev = valid ? __expf(c[r] - m8[r]) : 0.f;
                            l8[r] += ev;
                            pt[(r + 8 * hi) * 32 + sub * 16 + m15] = (half_t)ev;
                        }
                    } else {
#pragma unroll
                        for (int r = 0; r < 8; ++r)
                            pt[(r + 8 * hi) * 32 + sub * 16 + m15] = (half_t)0.f;
                    }
                }
                v16h ap  = load_a(pt, 32);
                v16h bv0 = load_b_cols(vbuf + (kt2 * 16) * D128 + h * HDIM, D128);
                v16h bv1 = load_b_cols(vbuf + (kt2 * 16) * D128 + h * HDIM + 16, D128);
                o0 = wmma32(ap, bv0, o0);
                o1 = wmma32(ap, bv1, o1);
            }
#pragma unroll
            for (int r = 0; r < 8; ++r) l8[r] = redsum16(l8[r]);
            // q columns of head h are dead now: store ctx there
#pragma unroll
            for (int r = 0; r < 8; ++r) {
                float inv = 1.f / l8[r];
                qb[(r + 8 * hi) * D128 + h * HDIM + m15]      = (half_t)(o0[r] * inv);
                qb[(r + 8 * hi) * D128 + h * HDIM + 16 + m15] = (half_t)(o1[r] * inv);
            }
        }
        // ---- out-proj + bias + residual(LN1) ; then LN2 ; store
        v8f accs[8];
#pragma unroll
        for (int n = 0; n < 8; ++n) {
            v8f acc = zero8();
#pragma unroll
            for (int kc = 0; kc < 4; ++kc) {
                v16h a  = load_a(qb + kc * 32, D128);
                v16h bm = load_b_rows(wos + (n * 16) * D128 + kc * 32, D128);
                acc = wmma32(a, bm, acc);
            }
            float bias = bos[n * 16 + m15];
#pragma unroll
            for (int r = 0; r < 8; ++r)
                acc[r] += bias + (float)ql[(r + 8 * hi) * D128 + n * 16 + m15];
            accs[n] = acc;
        }
#pragma unroll
        for (int r = 0; r < 8; ++r) {
            float s = 0.f;
#pragma unroll
            for (int n = 0; n < 8; ++n) s += accs[n][r];
            float mu = redsum16(s) * (1.f / 128.f);
            float vs = 0.f;
#pragma unroll
            for (int n = 0; n < 8; ++n) { float d = accs[n][r] - mu; vs += d * d; }
            float rstd = rsqrtf(redsum16(vs) * (1.f / 128.f) + 1e-8f);
            int R = r + 8 * hi, gr = r0 + R;
            bool ok = gr < L;
#pragma unroll
            for (int n = 0; n < 8; ++n) {
                int c = n * 16 + m15;
                float val = ok ? ((accs[n][r] - mu) * rstd * g2[c] + bb2[c]) : 0.f;
                og[gr * D128 + c] = (half_t)val;
            }
        }
    }
}

// ---------------------------------------------------------------- kernel 5: FFN + masked mean pool
__global__ __launch_bounds__(256) void k_ffn(
        const half_t* __restrict__ Oh,
        const float* __restrict__ w1, const float* __restrict__ fb1,
        const float* __restrict__ w2, const float* __restrict__ fb2,
        const int* __restrict__ lengths, float* __restrict__ out) {
    __shared__ __align__(16) half_t w1s[D128 * D128];
    __shared__ __align__(16) half_t w2s[D128 * D128];
    __shared__ __align__(16) half_t htile[8][16 * D128];
    __shared__ float b1s[D128], b2s[D128], pool[D128];

    int b = blockIdx.x, tid = threadIdx.x;
    for (int i = tid; i < D128 * D128; i += 256) {
        w1s[i] = (half_t)w1[i];
        w2s[i] = (half_t)w2[i];
    }
    if (tid < D128) { b1s[tid] = fb1[tid]; b2s[tid] = fb2[tid]; pool[tid] = 0.f; }
    __syncthreads();

    int L = lengths[b];
    int wave = tid >> 5, lane = tid & 31, m15 = lane & 15, hi = lane >> 4;
    const half_t* og = Oh + (size_t)b * LMAX * D128;

    for (int t = wave; t < LMAX / 16; t += 8) {
        int r0 = t * 16;
        if (r0 >= L) continue;
        const half_t* abase = og + r0 * D128;
        half_t* ht = htile[wave];
#pragma unroll
        for (int n = 0; n < 8; ++n) {
            v8f acc = zero8();
#pragma unroll
            for (int kc = 0; kc < 4; ++kc) {
                v16h a  = load_a(abase + kc * 32, D128);
                v16h bm = load_b_rows(w1s + (n * 16) * D128 + kc * 32, D128);
                acc = wmma32(a, bm, acc);
            }
            float bias = b1s[n * 16 + m15];
#pragma unroll
            for (int r = 0; r < 8; ++r)
                ht[(r + 8 * hi) * D128 + n * 16 + m15] =
                    (half_t)fmaxf(acc[r] + bias, 0.f);
        }
#pragma unroll
        for (int n = 0; n < 8; ++n) {
            v8f acc = zero8();
#pragma unroll
            for (int kc = 0; kc < 4; ++kc) {
                v16h a  = load_a(ht + kc * 32, D128);
                v16h bm = load_b_rows(w2s + (n * 16) * D128 + kc * 32, D128);
                acc = wmma32(a, bm, acc);
            }
            int c = n * 16 + m15;
            float bias = b2s[c];
#pragma unroll
            for (int r = 0; r < 8; ++r) {
                int R = r + 8 * hi, gr = r0 + R;
                if (gr < L) {
                    float val = acc[r] + bias + (float)abase[R * D128 + c];
                    atomicAdd(&pool[c], val);
                }
            }
        }
    }
    __syncthreads();
    if (tid < D128) out[(size_t)b * D128 + tid] = pool[tid] / (float)L;
}

// ---------------------------------------------------------------- launch

extern "C" void kernel_launch(void* const* d_in, const int* in_sizes, int n_in,
                              void* d_out, int out_size, void* d_ws, size_t ws_size,
                              hipStream_t stream) {
    const float* POI        = (const float*)d_in[0];
    const float* dde        = (const float*)d_in[1];
    const float* attW       = (const float*)d_in[2];
    const float* a_src      = (const float*)d_in[3];
    const float* a_dst      = (const float*)d_in[4];
    const float* in_proj_w  = (const float*)d_in[5];
    const float* in_proj_b  = (const float*)d_in[6];
    const float* out_proj_w = (const float*)d_in[7];
    const float* out_proj_b = (const float*)d_in[8];
    const float* ln1g       = (const float*)d_in[9];
    const float* ln1b       = (const float*)d_in[10];
    const float* ln2g       = (const float*)d_in[11];
    const float* ln2b       = (const float*)d_in[12];
    const float* ffn_w1     = (const float*)d_in[13];
    const float* ffn_b1     = (const float*)d_in[14];
    const float* ffn_w2     = (const float*)d_in[15];
    const float* ffn_b2     = (const float*)d_in[16];
    const int*   sess_idx   = (const int*)d_in[17];
    const int*   edge_dist  = (const int*)d_in[18];
    // d_in[19]=edge_index, d_in[20]=batch_ids, d_in[21]=node_pos (derivable, unused)
    const int*   lengths    = (const int*)d_in[22];
    int B = in_sizes[22];

    // workspace layout
    float* w_src  = (float*)d_ws;
    float* w_dst  = w_src + D128;
    int*   starts = (int*)(w_dst + D128);
    half_t* Hh = (half_t*)((char*)d_ws + 8192);
    size_t pe = (size_t)B * LMAX * D128;
    half_t* kh = Hh + pe;
    half_t* vh = kh + pe;
    half_t* Oh = vh + pe;
    float*  out = (float*)d_out;

    k_prep<<<1, 128, 0, stream>>>(attW, a_src, a_dst, lengths, w_src, w_dst, starts, B);
    k_gcn<<<B, 256, 0, stream>>>(POI, dde, sess_idx, edge_dist, lengths, starts,
                                 w_src, w_dst, Hh);
    int Mtiles = (int)(pe / (16 * D128));
    dim3 gkv((Mtiles + 7) / 8, 2);
    k_kv<<<gkv, 256, 0, stream>>>(Hh, in_proj_w, in_proj_b, kh, vh, Mtiles);
    k_attn<<<B, 256, 0, stream>>>(Hh, kh, vh, in_proj_w, in_proj_b,
                                  out_proj_w, out_proj_b,
                                  ln1g, ln1b, ln2g, ln2b, lengths, Oh);
    k_ffn<<<B, 256, 0, stream>>>(Oh, ffn_w1, ffn_b1, ffn_w2, ffn_b2, lengths, out);
}
